// Encoder_coarse_43568148251444
// MI455X (gfx1250) — compile-verified, matrix operand order checked by reading, not run on hardware
//
#include <hip/hip_runtime.h>

#define RES_PT 32
#define RES_R  8
#define BATCH  8
#define PI_F   3.14159265358979323846f

typedef __attribute__((ext_vector_type(16))) _Float16 v16h;
typedef __attribute__((ext_vector_type(8)))  float    v8f;

// order-preserving float<->uint encoding for atomic max
__device__ __forceinline__ unsigned f2ord(float f) {
  unsigned u = __float_as_uint(f);
  return (u & 0x80000000u) ? ~u : (u | 0x80000000u);
}
__device__ __forceinline__ float ord2f(unsigned u) {
  return (u & 0x80000000u) ? __uint_as_float(u & 0x7FFFFFFFu)
                           : __uint_as_float(~u);
}

// ---------------- init: zero grid + max accumulators ----------------
__global__ void k_init(float* grid, unsigned* pmax) {
  int i = blockIdx.x * blockDim.x + threadIdx.x;
  if (i < BATCH * RES_PT * RES_PT * RES_R) grid[i] = 0.f;
  if (i < BATCH * 32) pmax[i] = 0u;   // == encoded(-NaN), below encoded(-inf)
}

// ---------------- scatter: gaussian splat of w into (b,t,p,r) ----------------
__global__ void k_scatter(const float* __restrict__ pts,
                          const float* __restrict__ scp, const float* __restrict__ sgp,
                          const float* __restrict__ Ap,  const float* __restrict__ ivp,
                          float* __restrict__ grid, int npts, int n) {
  int i = blockIdx.x * blockDim.x + threadIdx.x;
  if (i >= npts) return;
  float scale = scp[0], sigma = sgp[0], A = Ap[0], iv = ivp[0];
  float x = pts[3*i], y = pts[3*i+1], z = pts[3*i+2];
  float r = fmaxf(sqrtf(x*x + y*y + z*z), 0.1f);
  float cz = fminf(fmaxf(z / r, -1.f), 1.f);
  float theta = acosf(cz);
  float phi = atan2f(y, x) + PI_F;
  float ct = theta * ((float)RES_PT / PI_F);
  float cp = phi * ((float)RES_PT / (2.f * PI_F));
  float cr = r / scale * (float)RES_R;
  int it = min(max((int)floorf(ct), 0), RES_PT - 1);
  int ip = min(max((int)floorf(cp), 0), RES_PT - 1);
  int ir = min(max((int)floorf(cr), 0), RES_R - 1);
  float dt = ct - ((float)it + iv);
  float dp = cp - ((float)ip + iv);
  float dr = cr - ((float)ir + iv);
  float d2 = dt*dt + dp*dp + dr*dr;
  float w = A * __expf(-d2 / (2.f * sigma * sigma));
  int b = i / n;
  atomicAdd(&grid[((b * RES_PT + it) * RES_PT + ip) * RES_R + ir], w);
}

// ---------------- conv1: (b,8,32,32) -> (b,32,32,32), 3x3 SAME, relu ----------------
__global__ void k_conv1(const float* __restrict__ grid,
                        const float* __restrict__ W, const float* __restrict__ bias,
                        float* __restrict__ h1) {
  int bb = blockIdx.x >> 5;
  int oc = blockIdx.x & 31;
  __shared__ float sw[8 * 9];
  __shared__ float sb;
  if (threadIdx.x < 72) sw[threadIdx.x] = W[oc * 72 + threadIdx.x];
  if (threadIdx.x == 0) sb = bias[oc];
  __syncthreads();
  for (int q = 0; q < 4; ++q) {
    int pos = threadIdx.x + q * 256;
    int x = pos >> 5, y = pos & 31;
    float acc = sb;
    for (int ic = 0; ic < 8; ++ic)
      for (int dx = 0; dx < 3; ++dx) {
        int xx = x + dx - 1;
        if ((unsigned)xx >= 32u) continue;
        for (int dy = 0; dy < 3; ++dy) {
          int yy = y + dy - 1;
          if ((unsigned)yy >= 32u) continue;
          // ft[b][r=ic][x][y] = grid[b][x][y][ic]
          acc += grid[((bb * 32 + xx) * 32 + yy) * 8 + ic] * sw[(ic * 3 + dx) * 3 + dy];
        }
      }
    h1[(size_t)(bb * 32 + oc) * 1024 + pos] = fmaxf(acc, 0.f);
  }
}

// ---------------- conv2 + quadrature reduction ----------------
// h2 (f16, pos-major) for WMMA, ftsum[b,c] = sum_{x,y} h2 * wq[y]
__global__ void k_conv2(const float* __restrict__ h1,
                        const float* __restrict__ W, const float* __restrict__ bias,
                        _Float16* __restrict__ h2, float* __restrict__ ftsum) {
  int bb = blockIdx.x >> 6;
  int oc = blockIdx.x & 63;
  __shared__ float sw[32 * 9];
  __shared__ float swq[32];
  __shared__ float red[256];
  int tid = threadIdx.x;
  for (int i = tid; i < 288; i += 256) sw[i] = W[oc * 288 + i];
  if (tid < 32) {   // Driscoll-Healy quadrature weights, bw = 16
    float s = 0.f;
    float a = (float)(2 * tid + 1) * (PI_F / 64.f);
    for (int k = 0; k < 16; ++k)
      s += __sinf(a * (float)(2 * k + 1)) / (float)(2 * k + 1);
    swq[tid] = 0.125f * __sinf(a) * s;
  }
  __syncthreads();
  float b0 = bias[oc];
  float wsum = 0.f;
  for (int q = 0; q < 4; ++q) {
    int pos = tid + q * 256;
    int x = pos >> 5, y = pos & 31;
    float acc = b0;
    for (int ic = 0; ic < 32; ++ic) {
      const float* hp = &h1[(size_t)(bb * 32 + ic) * 1024];
      for (int dx = 0; dx < 3; ++dx) {
        int xx = x + dx - 1;
        if ((unsigned)xx >= 32u) continue;
        for (int dy = 0; dy < 3; ++dy) {
          int yy = y + dy - 1;
          if ((unsigned)yy >= 32u) continue;
          acc += hp[xx * 32 + yy] * sw[(ic * 3 + dx) * 3 + dy];
        }
      }
    }
    acc = fmaxf(acc, 0.f);
    h2[((size_t)bb * 1024 + pos) * 64 + oc] = (_Float16)acc;
    wsum += acc * swq[y];
  }
  red[tid] = wsum;
  __syncthreads();
  for (int s = 128; s > 0; s >>= 1) {
    if (tid < s) red[tid] += red[tid + s];
    __syncthreads();
  }
  if (tid == 0) ftsum[bb * 64 + oc] = red[0];
}

// ---------------- bias2[b,d] = b_c3_1[d] + sum_c W_c3_1[d,64+c]*gsum[c]*ftsum[b,c] ----------------
__global__ void k_bias2(const float* __restrict__ Wg, const float* __restrict__ Wc1,
                        const float* __restrict__ b1, const float* __restrict__ ftsum,
                        float* __restrict__ bias2) {
  int bb = blockIdx.x;
  int d = threadIdx.x;       // 64 threads
  __shared__ float gsum[64];
  float g = 0.f;
  for (int k = 0; k < 32; ++k) g += Wg[d * 32 + k];
  gsum[d] = g;
  __syncthreads();
  float acc = b1[d];
  for (int c = 0; c < 64; ++c)
    acc += Wc1[d * 128 + 64 + c] * gsum[c] * ftsum[bb * 64 + c];
  bias2[bb * 64 + d] = acc;
}

// ---------------- WMMA head: per (b, m-chunk, g): relu(A@Bg + bias2) @ B2 + b_c3_2 ----------------
#define LDA 72   // padded stride (halfs); 36 uints -> conflict-free B128 frag loads

__global__ __launch_bounds__(256)
void k_head(const _Float16* __restrict__ h2, const float* __restrict__ Wg,
            const float* __restrict__ Wc1, const float* __restrict__ Wc2,
            const float* __restrict__ bc2, const float* __restrict__ bias2,
            float* __restrict__ out, unsigned* __restrict__ pmax) {
  int g  = blockIdx.x & 31;
  int mc = (blockIdx.x >> 5) & 7;
  int bb = blockIdx.x >> 8;
  __shared__ _Float16 sA[128 * LDA];   // A: 128 pos x 64 ch
  __shared__ _Float16 sB1[64 * LDA];   // Bg^T: [d][c]
  __shared__ _Float16 sB2[32 * LDA];   // Wc2:  [e][d] (native layout)
  __shared__ _Float16 sC[128 * LDA];   // stage-1 result (f16)
  __shared__ float sBias[64];
  __shared__ float sOb[32];
  __shared__ unsigned smax[32];
  int tid = threadIdx.x;

  // A chunk: 128 rows of h2 (contiguous, copy as uints into padded rows)
  {
    const unsigned* src = (const unsigned*)(h2 + ((size_t)bb * 1024 + (size_t)mc * 128) * 64);
    unsigned* dst = (unsigned*)sA;
    #pragma unroll
    for (int i = 0; i < 16; ++i) {
      int idx = tid + i * 256;           // 4096 uints
      int r = idx >> 5, cu = idx & 31;
      dst[r * (LDA / 2) + cu] = src[idx];
    }
  }
  // Bg^T[d][c] = W_gamma[c,g] * W_c3_1[d,c]
  #pragma unroll
  for (int i = 0; i < 16; ++i) {
    int idx = tid + i * 256;             // 4096
    int d = idx >> 6, c = idx & 63;
    sB1[d * LDA + c] = (_Float16)(Wg[c * 32 + g] * Wc1[d * 128 + c]);
  }
  // B2[e][d] = W_c3_2[e,d]
  #pragma unroll
  for (int i = 0; i < 8; ++i) {
    int idx = tid + i * 256;             // 2048
    int e = idx >> 6, d = idx & 63;
    sB2[e * LDA + d] = (_Float16)Wc2[idx];
  }
  if (tid < 64) sBias[tid] = bias2[bb * 64 + tid];
  if (tid < 32) { sOb[tid] = bc2[tid]; smax[tid] = 0u; }
  __syncthreads();

  int lane = tid & 31;
  int wave = tid >> 5;
  int lr = lane & 15;   // row (A) / col (B) selector
  int lh = lane >> 4;   // K-half selector (+8)
  int mt = wave;        // each wave owns one 16-row M tile

  // ---- stage 1: (128x64) @ (64x64), K split into two x32 WMMAs ----
  v16h afr[2];
  #pragma unroll
  for (int kk = 0; kk < 2; ++kk) {
    int r = mt * 16 + lr;
    #pragma unroll
    for (int j = 0; j < 8; ++j) {
      int kb = kk * 32 + ((j & 4) ? 16 : 0) + ((j & 3) << 1) + (lh << 3);
      afr[kk][2 * j]     = sA[r * LDA + kb];
      afr[kk][2 * j + 1] = sA[r * LDA + kb + 1];
    }
  }
  #pragma unroll
  for (int nt = 0; nt < 4; ++nt) {
    v8f acc = {};
    #pragma unroll
    for (int kk = 0; kk < 2; ++kk) {
      v16h bfr;
      int rb = nt * 16 + lr;
      #pragma unroll
      for (int j = 0; j < 8; ++j) {
        int kb = kk * 32 + ((j & 4) ? 16 : 0) + ((j & 3) << 1) + (lh << 3);
        bfr[2 * j]     = sB1[rb * LDA + kb];
        bfr[2 * j + 1] = sB1[rb * LDA + kb + 1];
      }
      acc = __builtin_amdgcn_wmma_f32_16x16x32_f16(false, afr[kk], false, bfr,
                                                   (short)0, acc, false, false);
    }
    #pragma unroll
    for (int r = 0; r < 8; ++r) {
      int m = mt * 16 + lh * 8 + r;
      int d = nt * 16 + lr;
      sC[m * LDA + d] = (_Float16)fmaxf(acc[r] + sBias[d], 0.f);
    }
  }
  __syncthreads();

  // ---- stage 2: (128x64) @ (64x32) ----
  #pragma unroll
  for (int kk = 0; kk < 2; ++kk) {
    int r = mt * 16 + lr;
    #pragma unroll
    for (int j = 0; j < 8; ++j) {
      int kb = kk * 32 + ((j & 4) ? 16 : 0) + ((j & 3) << 1) + (lh << 3);
      afr[kk][2 * j]     = sC[r * LDA + kb];
      afr[kk][2 * j + 1] = sC[r * LDA + kb + 1];
    }
  }
  #pragma unroll
  for (int nt = 0; nt < 2; ++nt) {
    v8f acc = {};
    #pragma unroll
    for (int kk = 0; kk < 2; ++kk) {
      v16h bfr;
      int rb = nt * 16 + lr;
      #pragma unroll
      for (int j = 0; j < 8; ++j) {
        int kb = kk * 32 + ((j & 4) ? 16 : 0) + ((j & 3) << 1) + (lh << 3);
        bfr[2 * j]     = sB2[rb * LDA + kb];
        bfr[2 * j + 1] = sB2[rb * LDA + kb + 1];
      }
      acc = __builtin_amdgcn_wmma_f32_16x16x32_f16(false, afr[kk], false, bfr,
                                                   (short)0, acc, false, false);
    }
    int e = nt * 16 + lr;
    float bv = sOb[e];
    unsigned lmax = 0u;
    #pragma unroll
    for (int r = 0; r < 8; ++r) {
      int p = mc * 128 + mt * 16 + lh * 8 + r;
      int x = p >> 5, y = p & 31;
      float v = acc[r] + bv;
      out[((((size_t)bb * 32 + e) * 32 + x) * 32 + y) * 32 + g] = v;
      unsigned env = f2ord(v);
      lmax = lmax > env ? lmax : env;
    }
    atomicMax(&smax[e], lmax);
  }
  __syncthreads();
  if (tid < 32) atomicMax(&pmax[bb * 32 + tid], smax[tid]);
}

// ---------------- finalize ft_g ----------------
__global__ void k_final(const unsigned* __restrict__ pmax, float* __restrict__ ftg) {
  int i = threadIdx.x;
  if (i < BATCH * 32) ftg[i] = ord2f(pmax[i]);
}

extern "C" void kernel_launch(void* const* d_in, const int* in_sizes, int n_in,
                              void* d_out, int out_size, void* d_ws, size_t ws_size,
                              hipStream_t stream) {
  const float* pts   = (const float*)d_in[0];
  const float* scale = (const float*)d_in[1];
  const float* sigma = (const float*)d_in[2];
  const float* Amp   = (const float*)d_in[3];
  const float* iv    = (const float*)d_in[4];
  const float* W1    = (const float*)d_in[7];
  const float* b1    = (const float*)d_in[8];
  const float* W2    = (const float*)d_in[9];
  const float* b2    = (const float*)d_in[10];
  const float* Wgm   = (const float*)d_in[11];
  const float* Wc1   = (const float*)d_in[12];
  const float* bc1   = (const float*)d_in[13];
  const float* Wc2   = (const float*)d_in[14];
  const float* bc2   = (const float*)d_in[15];

  char* ws = (char*)d_ws;
  float*    grid  = (float*)ws;                                  // 65536 f32
  float*    h1    = (float*)(ws + (256 << 10));                  // 262144 f32
  _Float16* h2    = (_Float16*)(ws + (256 << 10) + (1 << 20));   // 524288 f16
  float*    ftsum = (float*)(ws + (256 << 10) + (2 << 20));      // 512 f32
  float*    bias2 = (float*)(ws + (256 << 10) + (2 << 20) + 4096);
  unsigned* pmax  = (unsigned*)(ws + (256 << 10) + (2 << 20) + 8192);

  int npts = in_sizes[0] / 3;       // b*n = 262144
  int n = npts / BATCH;

  float* out = (float*)d_out;
  float* ftg = out + (size_t)BATCH * 32 * 32 * 32 * 32;

  k_init<<<256, 256, 0, stream>>>(grid, pmax);
  k_scatter<<<(npts + 255) / 256, 256, 0, stream>>>(pts, scale, sigma, Amp, iv,
                                                    grid, npts, n);
  k_conv1<<<BATCH * 32, 256, 0, stream>>>(grid, W1, b1, h1);
  k_conv2<<<BATCH * 64, 256, 0, stream>>>(h1, W2, b2, h2, ftsum);
  k_bias2<<<BATCH, 64, 0, stream>>>(Wgm, Wc1, bc1, ftsum, bias2);
  k_head<<<BATCH * 8 * 32, 256, 0, stream>>>(h2, Wgm, Wc1, Wc2, bc2, bias2, out, pmax);
  k_final<<<1, 256, 0, stream>>>(pmax, ftg);
}
